// Attention_76725295776457
// MI455X (gfx1250) — matrix operand order repaired and checked by reading, not compiled, not run on hardware
//
#include <hip/hip_runtime.h>
#include <hip/hip_bf16.h>

// ---------------------------------------------------------------------------
// Fused attention block for MI455X (gfx1250, wave32, WMMA).
//   x[8,1024,512] fp32 -> bf16 staging -> QKV GEMMs (v_wmma_f32_16x16x32_bf16,
//   B panel staged in LDS via double-buffered global_load_async_to_lds_b128)
//   -> flash-style attention per (b,h,16-row q-tile) -> LayerNorm -> fp32 out.
// ---------------------------------------------------------------------------

typedef __bf16 bf16_t;
typedef __attribute__((ext_vector_type(16))) __bf16 v16bf;
typedef __attribute__((ext_vector_type(8)))  float  v8f;
typedef __attribute__((ext_vector_type(4)))  unsigned int u32x4;

#define EMB   512
#define SEQ   1024
#define BATCH 8
#define NHEAD 8
#define HDIM  64
#define MROWS (BATCH * SEQ)   // 8192
#define GK    64              // K-slice per async stage in the GEMM

union FragU {
    v16bf v;
    u32x4 q[2];
};

// Load 16 bf16 values (two 16-byte chunks) into a WMMA fragment register set.
static __device__ inline v16bf load16(const bf16_t* p0, const bf16_t* p1) {
    FragU f;
    f.q[0] = *(const u32x4*)p0;
    f.q[1] = *(const u32x4*)p1;
    return f.v;
}

// ---------------------------------------------------------------------------
// fp32 -> bf16 conversion
// ---------------------------------------------------------------------------
__global__ void cvt_bf16(const float* __restrict__ src, bf16_t* __restrict__ dst, int n) {
    int i = blockIdx.x * blockDim.x + threadIdx.x;
    if (i < n) dst[i] = (bf16_t)src[i];
}

// ---------------------------------------------------------------------------
// QKV projection: Y[m,n] = sum_k X[m,k] * W[n,k]   (x @ W^T)
// Block = 128 rows x 64 cols (8 waves x 16 rows), sharing one 64-col W panel.
// W panel staged in LDS in 64-K slices, double-buffered, copied with the
// CDNA5 async-to-LDS path (ASYNCcnt) so the copy overlaps the WMMAs.
// Q,K stored [bh, s, d]; V stored transposed [bh, d, s] for the PV stage.
// ---------------------------------------------------------------------------
__global__ __launch_bounds__(256) void qkv_gemm(
    const bf16_t* __restrict__ xb,    // [8192, 512]
    const bf16_t* __restrict__ wb,    // [3, 512, 512]
    bf16_t* __restrict__ Qo,          // [64, 1024, 64]
    bf16_t* __restrict__ Ko,          // [64, 1024, 64]
    bf16_t* __restrict__ Vt)          // [64, 64, 1024]
{
    __shared__ bf16_t bsl[2][64 * GK];    // two 8 KB B slices (N=64 rows x K=64)

    const int tid  = threadIdx.x;
    const int lane = tid & 31;
    const int L2   = lane & 15;
    const int hi   = lane >> 4;
    const int wave = tid >> 5;

    const int mblk = blockIdx.x >> 3;     // 0..63  (128-row blocks)
    const int nblk = blockIdx.x & 7;      // 0..7   (64-col panels)
    const int m0   = mblk * 128 + wave * 16;
    const int n0   = nblk * 64;
    const int which = blockIdx.y;         // 0=Q, 1=K, 2=V

    const bf16_t* w    = wb + (size_t)which * EMB * EMB;
    const bf16_t* arow = xb + (size_t)(m0 + L2) * EMB;

    // Async staging: thread `tid` copies 32 bytes of the slice:
    //   row = tid/4 (N within panel), chunk = tid%4 (32-byte K chunk).
    const int srow = tid >> 2;
    const int schk = tid & 3;
    const bf16_t* wsrc = w + (size_t)(n0 + srow) * EMB + schk * 16;

    auto stage = [&](int buf, int k0) {
        const bf16_t* src = wsrc + k0;
        unsigned dst = (unsigned)(uintptr_t)&bsl[buf][srow * GK + schk * 16];
        asm volatile(
            "global_load_async_to_lds_b128 %0, %1, off\n\t"
            "global_load_async_to_lds_b128 %0, %1, off offset:16"
            :: "v"(dst), "v"(src) : "memory");
    };

    v8f zero = {};
    v8f acc[4];
    acc[0] = zero; acc[1] = zero; acc[2] = zero; acc[3] = zero;

    stage(0, 0);                          // prefetch first slice

    for (int k0 = 0; k0 < EMB; k0 += GK) {
        const int cur = (k0 >> 6) & 1;
        asm volatile("s_wait_asynccnt 0" ::: "memory");   // my slice landed
        __syncthreads();                                  // everyone's did
        if (k0 + GK < EMB) stage(cur ^ 1, k0 + GK);       // overlap next copy

        #pragma unroll
        for (int kk = 0; kk < GK; kk += 32) {
            // A fragment (16x32 bf16) streamed from global per wave.
            v16bf a = load16(arow + k0 + kk + hi * 8,
                             arow + k0 + kk + 16 + hi * 8);
            // B fragments from LDS (8x cross-wave reuse).
            #pragma unroll
            for (int t = 0; t < 4; ++t) {
                const bf16_t* bp = &bsl[cur][(t * 16 + L2) * GK + kk];
                v16bf b = load16(bp + hi * 8, bp + 16 + hi * 8);
                acc[t] = __builtin_amdgcn_wmma_f32_16x16x32_bf16(
                    false, a, false, b, (short)0, acc[t], false, false);
            }
        }
    }

    // C layout: VGPR g, lanes 0-15 -> (M=g, N=L2); lanes 16-31 -> (M=g+8, N=L2).
    #pragma unroll
    for (int t = 0; t < 4; ++t) {
        #pragma unroll
        for (int g = 0; g < 8; ++g) {
            int row = m0 + g + 8 * hi;          // global M (0..8191)
            int col = n0 + t * 16 + L2;         // global N (0..511)
            int b_  = row >> 10;
            int s   = row & 1023;
            int h   = col >> 6;
            int d   = col & 63;
            int bh  = b_ * NHEAD + h;
            bf16_t val = (bf16_t)acc[t][g];
            if (which == 0)      Qo[((size_t)bh * SEQ + s) * HDIM + d] = val;
            else if (which == 1) Ko[((size_t)bh * SEQ + s) * HDIM + d] = val;
            else                 Vt[((size_t)bh * HDIM + d) * SEQ + s] = val;
        }
    }
}

// ---------------------------------------------------------------------------
// Flash attention: one wave per (bh, 16-query tile); loops over 32-key chunks.
// Scores 16x32 via 2x2 WMMA; online softmax with 16-lane shuffle reductions;
// P round-trips through LDS to convert C-layout -> A-layout; PV via 4 WMMAs.
// V fragments are loaded before the softmax chain so their latency hides
// under the exp/VALU work; next chunk is prefetched a full iteration early.
// ---------------------------------------------------------------------------
__global__ __launch_bounds__(32) void attn_fused(
    const bf16_t* __restrict__ Q, const bf16_t* __restrict__ K,
    const bf16_t* __restrict__ Vt, float* __restrict__ O)
{
    __shared__ bf16_t pbuf[16 * 32];

    const int lane = threadIdx.x & 31;
    const int L2   = lane & 15;
    const int hi   = lane >> 4;
    const int bh   = blockIdx.y;       // 0..63
    const int q0   = blockIdx.x * 16;  // query tile start
    const float scale = 0.04419417382415922f;   // 512^-0.5

    // Q tile A-fragments (held for the whole loop): 16 rows x 64 d.
    const bf16_t* qrow = Q + ((size_t)bh * SEQ + q0 + L2) * HDIM;
    v16bf aq0 = load16(qrow +  0 + hi * 8, qrow + 16 + hi * 8);
    v16bf aq1 = load16(qrow + 32 + hi * 8, qrow + 48 + hi * 8);

    const bf16_t* kbase = K  + ((size_t)bh * SEQ + L2) * HDIM;
    const bf16_t* vbase = Vt + ((size_t)bh * HDIM + L2) * SEQ;

    float mrow[8], lrow[8];
    v8f zero = {};
    v8f acc[4];
    #pragma unroll
    for (int g = 0; g < 8; ++g) { mrow[g] = -3.0e38f; lrow[g] = 0.0f; }
    acc[0] = zero; acc[1] = zero; acc[2] = zero; acc[3] = zero;

    for (int j0 = 0; j0 < SEQ; j0 += 32) {
        // Prefetch next chunk's K and V lines toward the caches.
        if (j0 + 32 < SEQ) {
            __builtin_prefetch(kbase + (size_t)(j0 + 32) * HDIM, 0, 1);
            __builtin_prefetch(vbase + j0 + 32, 0, 1);
        }

        // ---- S = Q K^T for 32 keys (two 16x16 C tiles) ----
        v8f c[2];
        #pragma unroll
        for (int jt = 0; jt < 2; ++jt) {
            const bf16_t* krow = kbase + (size_t)(j0 + jt * 16) * HDIM;
            v16bf b0 = load16(krow +  0 + hi * 8, krow + 16 + hi * 8);
            v16bf b1 = load16(krow + 32 + hi * 8, krow + 48 + hi * 8);
            v8f s = zero;
            s = __builtin_amdgcn_wmma_f32_16x16x32_bf16(false, aq0, false, b0, (short)0, s, false, false);
            s = __builtin_amdgcn_wmma_f32_16x16x32_bf16(false, aq1, false, b1, (short)0, s, false, false);
            c[jt] = s;
        }

        // ---- V fragments issued early (independent of softmax chain) ----
        v16bf bv[4];
        #pragma unroll
        for (int t = 0; t < 4; ++t) {
            const bf16_t* vrow = vbase + (size_t)(t * 16) * SEQ + j0;
            bv[t] = load16(vrow + hi * 8, vrow + 16 + hi * 8);
        }

        // ---- online softmax (row M = g + 8*hi lives on one 16-lane group) ----
        float p0[8], p1[8];
        #pragma unroll
        for (int g = 0; g < 8; ++g) {
            float s0 = c[0][g] * scale;
            float s1 = c[1][g] * scale;
            float rmax = fmaxf(s0, s1);
            #pragma unroll
            for (int m = 1; m <= 8; m <<= 1)
                rmax = fmaxf(rmax, __shfl_xor(rmax, m, 32));
            float mnew = fmaxf(mrow[g], rmax);
            float fac  = __expf(mrow[g] - mnew);
            float e0   = __expf(s0 - mnew);
            float e1   = __expf(s1 - mnew);
            float rsum = e0 + e1;
            #pragma unroll
            for (int m = 1; m <= 8; m <<= 1)
                rsum += __shfl_xor(rsum, m, 32);
            lrow[g] = lrow[g] * fac + rsum;
            mrow[g] = mnew;
            p0[g] = e0;
            p1[g] = e1;
            #pragma unroll
            for (int t = 0; t < 4; ++t) acc[t][g] *= fac;
        }

        // ---- C-layout -> A-layout via LDS (16 rows x 32 key cols, bf16) ----
        #pragma unroll
        for (int g = 0; g < 8; ++g) {
            int row = g + 8 * hi;
            pbuf[row * 32 + L2]      = (bf16_t)p0[g];
            pbuf[row * 32 + 16 + L2] = (bf16_t)p1[g];
        }
        __syncthreads();
        const bf16_t* prow = pbuf + L2 * 32;
        v16bf ap = load16(prow + hi * 8, prow + 16 + hi * 8);
        __syncthreads();

        // ---- O += P @ V ----
        #pragma unroll
        for (int t = 0; t < 4; ++t) {
            acc[t] = __builtin_amdgcn_wmma_f32_16x16x32_bf16(
                false, ap, false, bv[t], (short)0, acc[t], false, false);
        }
    }

    // ---- epilogue: O /= l, scatter into [b, s, h*64 + d] fp32 ----
    const int b_ = bh >> 3;
    const int h  = bh & 7;
    #pragma unroll
    for (int t = 0; t < 4; ++t) {
        #pragma unroll
        for (int g = 0; g < 8; ++g) {
            int row = q0 + g + 8 * hi;
            int col = h * HDIM + t * 16 + L2;
            O[((size_t)b_ * SEQ + row) * EMB + col] = acc[t][g] / lrow[g];
        }
    }
}

// ---------------------------------------------------------------------------
// LayerNorm over the last dim (512), one block per row.
// ---------------------------------------------------------------------------
__global__ __launch_bounds__(256) void layernorm_k(
    const float* __restrict__ O, const float* __restrict__ gamma,
    const float* __restrict__ beta, float* __restrict__ out)
{
    __shared__ float ssum[256];
    __shared__ float ssq[256];
    const int row = blockIdx.x;
    const int tid = threadIdx.x;
    const float* r = O + (size_t)row * EMB;
    float v0 = r[tid];
    float v1 = r[tid + 256];
    ssum[tid] = v0 + v1;
    ssq[tid]  = v0 * v0 + v1 * v1;
    __syncthreads();
    #pragma unroll
    for (int w = 128; w > 0; w >>= 1) {
        if (tid < w) { ssum[tid] += ssum[tid + w]; ssq[tid] += ssq[tid + w]; }
        __syncthreads();
    }
    float mu   = ssum[0] * (1.0f / EMB);
    float var  = ssq[0] * (1.0f / EMB) - mu * mu;
    float rstd = rsqrtf(var + 1e-5f);
    float* o = out + (size_t)row * EMB;
    o[tid]       = (v0 - mu) * rstd * gamma[tid]       + beta[tid];
    o[tid + 256] = (v1 - mu) * rstd * gamma[tid + 256] + beta[tid + 256];
}

// ---------------------------------------------------------------------------
// Workspace layout (bytes):
//   xb   @ 0          : 8192*512 bf16  = 8,388,608
//   wb   @ 8,388,608  : 3*512*512 bf16 = 1,572,864
//   Qb   @ 9,961,472  : 8,388,608
//   Kb   @ 18,350,080 : 8,388,608
//   Vtb  @ 26,738,688 : 8,388,608
//   Ob   @ 35,127,296 : 8192*512 f32   = 16,777,216   (end ~51.9 MB)
// ---------------------------------------------------------------------------
extern "C" void kernel_launch(void* const* d_in, const int* in_sizes, int n_in,
                              void* d_out, int out_size, void* d_ws, size_t ws_size,
                              hipStream_t stream) {
    const float* x     = (const float*)d_in[0];
    const float* Wq    = (const float*)d_in[1];
    const float* Wk    = (const float*)d_in[2];
    const float* Wv    = (const float*)d_in[3];
    const float* gamma = (const float*)d_in[4];
    const float* beta  = (const float*)d_in[5];

    char* ws = (char*)d_ws;
    bf16_t* xb  = (bf16_t*)(ws + 0);
    bf16_t* wb  = (bf16_t*)(ws + 8388608);
    bf16_t* Qb  = (bf16_t*)(ws + 9961472);
    bf16_t* Kb  = (bf16_t*)(ws + 18350080);
    bf16_t* Vtb = (bf16_t*)(ws + 26738688);
    float*  Ob  = (float*)(ws + 35127296);

    const int nx = MROWS * EMB;        // 4,194,304
    const int nw = EMB * EMB;          // 262,144
    cvt_bf16<<<(nx + 255) / 256, 256, 0, stream>>>(x,  xb,          nx);
    cvt_bf16<<<(nw + 255) / 256, 256, 0, stream>>>(Wq, wb,          nw);
    cvt_bf16<<<(nw + 255) / 256, 256, 0, stream>>>(Wk, wb + nw,     nw);
    cvt_bf16<<<(nw + 255) / 256, 256, 0, stream>>>(Wv, wb + 2 * nw, nw);

    dim3 ggrid(512, 3);                // 64 row-blocks x 8 col-panels, 3 matrices
    qkv_gemm<<<ggrid, 256, 0, stream>>>(xb, wb, Qb, Kb, Vtb);

    dim3 agrid(SEQ / 16, BATCH * NHEAD);   // (64, 64)
    attn_fused<<<agrid, 32, 0, stream>>>(Qb, Kb, Vtb, Ob);

    layernorm_k<<<MROWS, 256, 0, stream>>>(Ob, gamma, beta, (float*)d_out);
}
